// InteractionPPBlock_5952824672736
// MI455X (gfx1250) — matrix operand order repaired and checked by reading, not compiled, not run on hardware
//
#include <hip/hip_runtime.h>
#include <hip/hip_bf16.h>

// ---------------------------------------------------------------------------
// DimeNet++ InteractionPPBlock for MI455X (gfx1250, wave32).
//  - All big GEMMs: v_wmma_f32_16x16x32_f16 (f16 in, f32 accumulate).
//  - Weight panels -> LDS via Tensor Data Mover (tensor_load_to_lds +
//    s_wait_tensorcnt), overlapped with activation staging.
//  - Triplet scatter-mean: L2-resident gather + global_atomic_add_f32.
// Workload is memory-bound (~2 GB @ 23.3 TB/s >> 95 GFLOP of matrix math),
// so activations stay f32 in HBM; WMMA keeps compute off the critical path.
// ---------------------------------------------------------------------------

typedef _Float16 half_t;
typedef __attribute__((ext_vector_type(16))) _Float16 v16h;
typedef __attribute__((ext_vector_type(8)))  float    v8f;
typedef __attribute__((ext_vector_type(4)))  unsigned int u32x4;
typedef __attribute__((ext_vector_type(8)))  int      i32x8;
typedef __attribute__((ext_vector_type(4)))  int      i32x4;

constexpr int E_   = 300000;
constexpr int T_   = 3000000;
constexpr int HID  = 128;
constexpr int INTD = 64;

// ---- workspace layout (bytes) ----
constexpr size_t OFF_XJI  = 0;                                   // [E,128] f32
constexpr size_t OFF_XKD  = OFF_XJI  + (size_t)E_ * HID  * 4;    // [E,64]  f32
constexpr size_t OFF_SSUM = OFF_XKD  + (size_t)E_ * INTD * 4;    // [E,64]  f32
constexpr size_t OFF_CNT  = OFF_SSUM + (size_t)E_ * INTD * 4;    // [E]     f32
constexpr size_t OFF_HBUF = OFF_CNT  + (size_t)E_ * 4;           // [E,128] f32
constexpr size_t OFF_WR   = OFF_HBUF + (size_t)E_ * HID * 4;     // [6,128] f32
constexpr size_t OFF_F16  = OFF_WR   + 6 * 128 * 4;
constexpr size_t H16      = 128 * 128 * 2;                       // 32 KB each
constexpr size_t OFF_WJIT  = OFF_F16 + 0 * H16;
constexpr size_t OFF_WKJT  = OFF_F16 + 1 * H16;
constexpr size_t OFF_WLINT = OFF_F16 + 2 * H16;
constexpr size_t OFF_W1BT  = OFF_F16 + 3 * H16;
constexpr size_t OFF_W2BT  = OFF_F16 + 4 * H16;
constexpr size_t OFF_W1A0T = OFF_F16 + 5 * H16;
constexpr size_t OFF_W2A0T = OFF_F16 + 6 * H16;
constexpr size_t OFF_W1A1T = OFF_F16 + 7 * H16;
constexpr size_t OFF_W2A1T = OFF_F16 + 8 * H16;
constexpr size_t OFF_WDOWNT= OFF_F16 + 9 * H16;                  // [64n][128k]
constexpr size_t OFF_WUPT  = OFF_WDOWNT + 64 * 128 * 2;          // [128n][64k]
constexpr size_t OFF_WSBFT = OFF_WUPT   + 128 * 64 * 2;          // [64n][64k]

__device__ __forceinline__ float silu_f(float v) {
    return v * (1.0f / (1.0f + __expf(-v)));
}

__device__ __forceinline__ v8f wmma_f16(v16h a, v16h b, v8f c) {
    return __builtin_amdgcn_wmma_f32_16x16x32_f16(
        false, a, false, b, (short)0, c, false, false);
}

// ---------------------------------------------------------------------------
// Tensor Data Mover: 1-D panel copy global -> LDS, ndwords 32-bit words.
// D# per CDNA5 ISA ch.8:
//   g0 = {count=1 | is_restore=0, lds_addr, global_addr[31:0],
//         global_addr[56:32] | type=2<<30}
//   g1 = {data_size=2(4B), tensor_dim0=tile_dim0=ndwords, tensor_dim1=1,
//         tile_dim1=1, dim0_stride=ndwords}
// This toolchain exposes the 6-arg builtin:
//   (u32x4 g0, i32x8 g1, i32x4 g2, i32x4 g3, i32x8 g4, i32 cpol)
// Issue from ONE wave per workgroup (EXEC ignored by TDM), drain with
// s_wait_tensorcnt 0, then workgroup barrier for visibility.
// ---------------------------------------------------------------------------
__device__ __forceinline__ void tdm_load_1d(const void* gsrc, void* lds_dst,
                                            unsigned ndwords) {
    unsigned long long ga = (unsigned long long)gsrc;
    unsigned lds = (unsigned)(unsigned long long)lds_dst;  // low 32b = LDS offset
    u32x4 g0 = { 1u,                                   // count=1
                 lds,
                 (unsigned)(ga & 0xFFFFFFFFu),
                 (unsigned)((ga >> 32) & 0x01FFFFFFu) | (2u << 30) };
    i32x8 g1 = { (int)(2u << 16),                      // data_size = 4B
                 (int)((ndwords & 0xFFFFu) << 16),     // tensor_dim0[15:0]
                 (int)(((ndwords >> 16) & 0xFFFFu) | (1u << 16)), // td0 hi | td1 lo
                 (int)((ndwords & 0xFFFFu) << 16),     // td1 hi=0 | tile_dim0
                 (int)1,                               // tile_dim1=1, tile_dim2=0
                 (int)ndwords,                         // dim0_stride[31:0]
                 0, 0 };
    i32x4 gz4 = { 0, 0, 0, 0 };
    i32x8 gz8 = { 0, 0, 0, 0, 0, 0, 0, 0 };
    __builtin_amdgcn_tensor_load_to_lds(g0, g1, gz4, gz4, gz8, 0);
}

__device__ __forceinline__ void tdm_drain() {
    __builtin_amdgcn_s_wait_tensorcnt(0);
}

// A fragment: 16xK f16 row tile in LDS (rowstride halves per row), block kb of 32 K.
// Layout: lanes 0-15 -> M=0..15, VGPR w holds K = 16*hf + 2w (matches B fragment).
__device__ __forceinline__ v16h load_a_frag(const half_t* rowbuf, int rowstride,
                                            int lane, int kb) {
    const int mrow = lane & 15, hf = lane >> 4;
    const half_t* rb = rowbuf + mrow * rowstride + kb * 32;
    union { v16h h; unsigned int u[8]; } r;
#pragma unroll
    for (int w = 0; w < 8; ++w) {
        const int kbase = 16 * hf + 2 * w;
        r.u[w] = *(const unsigned int*)(rb + kbase);
    }
    return r.h;
}

// B fragment: weights stored transposed [N][K] f16; lane -> column n = 16j+ncol,
// lanes 0-15 hold K = 2w (+kb*32), lanes 16-31 hold K = 16+2w (+kb*32).
__device__ __forceinline__ v16h load_b_frag(const half_t* wt, int kstride,
                                            int j, int kb, int lane) {
    const int ncol = lane & 15, hf = lane >> 4;
    const half_t* p = wt + (size_t)(j * 16 + ncol) * kstride + kb * 32 + hf * 16;
    union { v16h h; unsigned int u[8]; } r;
#pragma unroll
    for (int w = 0; w < 8; ++w) r.u[w] = *(const unsigned int*)(p + 2 * w);
    return r.h;
}

// ---------------------------------------------------------------------------
// Prep kernels
// ---------------------------------------------------------------------------
__global__ void k_wr(const float* __restrict__ Wrbf1, const float* __restrict__ Wrbf2,
                     float* __restrict__ Wr) {
    for (int i = threadIdx.x; i < 6 * 128; i += blockDim.x) {
        int r = i / 128, n = i % 128;
        float s = 0.f;
#pragma unroll
        for (int b = 0; b < 8; ++b) s += Wrbf1[r * 8 + b] * Wrbf2[b * 128 + n];
        Wr[i] = s;
    }
}

__global__ void k_wsbf(const float* __restrict__ Wsbf1, const float* __restrict__ Wsbf2,
                       half_t* __restrict__ Wsbft) {
    int g0 = blockIdx.x * blockDim.x + threadIdx.x;
    for (int i = g0; i < 64 * 64; i += gridDim.x * blockDim.x) {
        int n = i / 64, k = i % 64;
        float s = 0.f;
        if (k < 42) {
#pragma unroll
            for (int b = 0; b < 8; ++b) s += Wsbf1[k * 8 + b] * Wsbf2[b * 64 + n];
        }
        Wsbft[(size_t)n * 64 + k] = (half_t)s;
    }
}

// W row-major [K][N] f32 -> Wt [N][K] f16
__global__ void k_transpose(const float* __restrict__ W, half_t* __restrict__ Wt,
                            int K, int N) {
    int g0 = blockIdx.x * blockDim.x + threadIdx.x;
    int total = K * N;
    for (int i = g0; i < total; i += gridDim.x * blockDim.x) {
        int n = i / K, k = i % K;
        Wt[i] = (half_t)W[(size_t)k * N + n];
    }
}

__global__ void k_zero(float* __restrict__ p, long long n) {
    long long g = (long long)blockIdx.x * blockDim.x + threadIdx.x;
    long long stride = (long long)gridDim.x * blockDim.x;
    for (; g < n; g += stride) p[g] = 0.f;
}

// ---------------------------------------------------------------------------
// Edge kernel: x_ji = silu(x@Wji+bji); x_kj = silu(x@Wkj+bkj)*rbf_e;
//              xkd  = silu(x_kj@Wdown).   4 waves x 16 rows = 64 rows / block.
// ---------------------------------------------------------------------------
__global__ void __launch_bounds__(128) k_edge(
    const float* __restrict__ x, const float* __restrict__ rbf,
    const half_t* __restrict__ Wjit, const half_t* __restrict__ Wkjt,
    const half_t* __restrict__ Wdownt, const float* __restrict__ Wr,
    const float* __restrict__ bji, const float* __restrict__ bkj,
    float* __restrict__ xji_out, float* __restrict__ xkd_out) {
    __shared__ half_t sW[128 * 128];       // 32 KB (Wji -> Wkj -> Wdown, via TDM)
    __shared__ float  sWr[6 * 128];        // 3 KB
    __shared__ float  sRbf[4][16 * 6];     // 1.5 KB
    __shared__ half_t sA[4][16 * 128];     // 16 KB (x tile -> x_kj tile)

    const int tid = threadIdx.x, lane = tid & 31, wave = tid >> 5;
    const int hf = lane >> 4, ncol = lane & 15;
    const int rowbase = blockIdx.x * 64 + wave * 16;

    if (wave == 0) tdm_load_1d(Wjit, sW, 8192);        // 32 KB panel via TDM
    for (int i = tid; i < 768; i += 128) sWr[i] = Wr[i];

    // stage x tile (f32 -> f16 LDS) + rbf tile (overlaps the TDM)
    { const float4* x4 = (const float4*)x;
#pragma unroll
      for (int it = 0; it < 16; ++it) {
          int fidx = it * 32 + lane;
          int r = fidx >> 5, c4 = fidx & 31;
          int g = rowbase + r;
          float4 v = (g < E_) ? x4[(size_t)g * 32 + c4] : make_float4(0.f, 0.f, 0.f, 0.f);
          union { half_t h[4]; uint2 u; } pk;
          pk.h[0] = (half_t)v.x; pk.h[1] = (half_t)v.y;
          pk.h[2] = (half_t)v.z; pk.h[3] = (half_t)v.w;
          *(uint2*)&sA[wave][r * 128 + c4 * 4] = pk.u;
      }
      for (int q = lane; q < 96; q += 32) {
          int r = q / 6, c = q % 6;
          int g = rowbase + r;
          sRbf[wave][q] = (g < E_) ? rbf[(size_t)g * 6 + c] : 0.f;
      } }
    if (wave == 0) tdm_drain();
    __syncthreads();

    v16h a[4];
#pragma unroll
    for (int kb = 0; kb < 4; ++kb) a[kb] = load_a_frag(sA[wave], 128, lane, kb);

    // GEMM1: x @ Wji
    v8f acc[8] = {};
#pragma unroll
    for (int kb = 0; kb < 4; ++kb)
#pragma unroll
        for (int j = 0; j < 8; ++j)
            acc[j] = wmma_f16(a[kb], load_b_frag(sW, 128, j, kb, lane), acc[j]);

#pragma unroll
    for (int j = 0; j < 8; ++j) {
        const int n = j * 16 + ncol;
        const float bn = bji[n];
#pragma unroll
        for (int i = 0; i < 8; ++i) {
            int g = rowbase + i + 8 * hf;
            if (g < E_) xji_out[(size_t)g * 128 + n] = silu_f(acc[j][i] + bn);
        }
    }
    __syncthreads();
    if (wave == 0) { tdm_load_1d(Wkjt, sW, 8192); tdm_drain(); }
    __syncthreads();

    // GEMM2: x @ Wkj (A frags reused)
    v8f acc2[8] = {};
#pragma unroll
    for (int kb = 0; kb < 4; ++kb)
#pragma unroll
        for (int j = 0; j < 8; ++j)
            acc2[j] = wmma_f16(a[kb], load_b_frag(sW, 128, j, kb, lane), acc2[j]);

    // bias + silu + rbf_e scaling; stage x_kj back into sA as f16
#pragma unroll
    for (int j = 0; j < 8; ++j) {
        const int n = j * 16 + ncol;
        const float bn = bkj[n];
#pragma unroll
        for (int i = 0; i < 8; ++i) {
            const int m = i + 8 * hf;
            float v = silu_f(acc2[j][i] + bn);
            float re = 0.f;
#pragma unroll
            for (int r = 0; r < 6; ++r) re += sRbf[wave][m * 6 + r] * sWr[r * 128 + n];
            sA[wave][m * 128 + n] = (half_t)(v * re);
        }
    }
    __syncthreads();
    if (wave == 0) { tdm_load_1d(Wdownt, sW, 4096); tdm_drain(); }
    __syncthreads();

    // GEMM3: x_kj @ Wdown (N=64)
#pragma unroll
    for (int kb = 0; kb < 4; ++kb) a[kb] = load_a_frag(sA[wave], 128, lane, kb);
    v8f acc3[4] = {};
#pragma unroll
    for (int kb = 0; kb < 4; ++kb)
#pragma unroll
        for (int j = 0; j < 4; ++j)
            acc3[j] = wmma_f16(a[kb], load_b_frag(sW, 128, j, kb, lane), acc3[j]);

#pragma unroll
    for (int j = 0; j < 4; ++j) {
        const int n = j * 16 + ncol;
#pragma unroll
        for (int i = 0; i < 8; ++i) {
            int g = rowbase + i + 8 * hf;
            if (g < E_) xkd_out[(size_t)g * 64 + n] = silu_f(acc3[j][i]);
        }
    }
}

// ---------------------------------------------------------------------------
// Triplet kernel: sbf_t = sbfP @ WsbfP (WMMA, K padded 42->64);
// msg = xkd[idx_kj] * sbf_t; atomic scatter-add into ssum[idx_ji], cnt.
// ---------------------------------------------------------------------------
__global__ void __launch_bounds__(128) k_triplet(
    const float* __restrict__ sbf, const int* __restrict__ idx_kj,
    const int* __restrict__ idx_ji, const half_t* __restrict__ Wsbft,
    const float* __restrict__ xkd, float* __restrict__ ssum,
    float* __restrict__ cnt) {
    __shared__ half_t sWb[64 * 64];     // 8 KB (via TDM)
    __shared__ half_t sA[4][16 * 64];   // 8 KB

    const int tid = threadIdx.x, lane = tid & 31, wave = tid >> 5;
    const int hf = lane >> 4, ncol = lane & 15;
    const int tbase = blockIdx.x * 64 + wave * 16;

    if (wave == 0) tdm_load_1d(Wsbft, sWb, 2048);

    // stage sbf rows (42 wide, pad to 64) as f16 (overlaps the TDM)
    for (int q = lane; q < 16 * 64; q += 32) {
        int r = q >> 6, c = q & 63;
        long long t = (long long)tbase + r;
        float v = (c < 42 && t < T_) ? sbf[(size_t)t * 42 + c] : 0.f;
        sA[wave][q] = (half_t)v;
    }
    if (wave == 0) tdm_drain();
    __syncthreads();

    v16h a[2];
#pragma unroll
    for (int kb = 0; kb < 2; ++kb) a[kb] = load_a_frag(sA[wave], 64, lane, kb);
    v8f acc[4] = {};
#pragma unroll
    for (int kb = 0; kb < 2; ++kb)
#pragma unroll
        for (int j = 0; j < 4; ++j)
            acc[j] = wmma_f16(a[kb], load_b_frag(sWb, 64, j, kb, lane), acc[j]);

    int kjr[8], jir[8];
#pragma unroll
    for (int i = 0; i < 8; ++i) {
        long long t = (long long)tbase + i + 8 * hf;
        kjr[i] = idx_kj[t];
        jir[i] = idx_ji[t];
        __builtin_prefetch(&xkd[(size_t)kjr[i] * 64], 0, 3);  // global_prefetch_b8
    }

#pragma unroll
    for (int j = 0; j < 4; ++j) {
        const int n = j * 16 + ncol;
#pragma unroll
        for (int i = 0; i < 8; ++i) {
            float v = acc[j][i] * xkd[(size_t)kjr[i] * 64 + n];
            atomicAdd(&ssum[(size_t)jir[i] * 64 + n], v);
        }
    }
    if (lane < 16) {
        long long t = (long long)tbase + lane;
        atomicAdd(&cnt[idx_ji[t]], 1.0f);
    }
}

// ---------------------------------------------------------------------------
// Up kernel: mean = ssum/max(cnt,1); h = x_ji + silu(mean @ Wup)   -> d_out
// ---------------------------------------------------------------------------
__global__ void __launch_bounds__(128) k_up(
    const float* __restrict__ ssum, const float* __restrict__ cnt,
    const half_t* __restrict__ Wupt, const float* __restrict__ xji,
    float* __restrict__ hout) {
    __shared__ half_t sW[128 * 64];    // 16 KB, [128n][64k] (via TDM)
    __shared__ half_t sA[4][16 * 64];  // 8 KB

    const int tid = threadIdx.x, lane = tid & 31, wave = tid >> 5;
    const int hf = lane >> 4, ncol = lane & 15;
    const int rowbase = blockIdx.x * 64 + wave * 16;

    if (wave == 0) tdm_load_1d(Wupt, sW, 4096);

    { const float4* s4 = (const float4*)ssum;
#pragma unroll
      for (int it = 0; it < 8; ++it) {
          int fidx = it * 32 + lane;
          int r = fidx >> 4, c4 = fidx & 15;
          int g = rowbase + r;
          float inv = 0.f;
          float4 v = make_float4(0.f, 0.f, 0.f, 0.f);
          if (g < E_) {
              inv = 1.0f / fmaxf(cnt[g], 1.0f);
              v = s4[(size_t)g * 16 + c4];
          }
          union { half_t h[4]; uint2 u; } pk;
          pk.h[0] = (half_t)(v.x * inv); pk.h[1] = (half_t)(v.y * inv);
          pk.h[2] = (half_t)(v.z * inv); pk.h[3] = (half_t)(v.w * inv);
          *(uint2*)&sA[wave][r * 64 + c4 * 4] = pk.u;
      } }
    if (wave == 0) tdm_drain();
    __syncthreads();

    v16h a[2];
#pragma unroll
    for (int kb = 0; kb < 2; ++kb) a[kb] = load_a_frag(sA[wave], 64, lane, kb);
    v8f acc[8] = {};
#pragma unroll
    for (int kb = 0; kb < 2; ++kb)
#pragma unroll
        for (int j = 0; j < 8; ++j)
            acc[j] = wmma_f16(a[kb], load_b_frag(sW, 64, j, kb, lane), acc[j]);

#pragma unroll
    for (int j = 0; j < 8; ++j) {
        const int n = j * 16 + ncol;
#pragma unroll
        for (int i = 0; i < 8; ++i) {
            int g = rowbase + i + 8 * hf;
            if (g < E_)
                hout[(size_t)g * 128 + n] = xji[(size_t)g * 128 + n] + silu_f(acc[j][i]);
        }
    }
}

// ---------------------------------------------------------------------------
// Residual layer: hout = hin + silu(silu(hin@W1+b1)@W2+b2)
// ---------------------------------------------------------------------------
__global__ void __launch_bounds__(128) k_res(
    const float* __restrict__ hin, const half_t* __restrict__ W1t,
    const float* __restrict__ b1, const half_t* __restrict__ W2t,
    const float* __restrict__ b2, float* __restrict__ hout) {
    __shared__ half_t sW[128 * 128];   // 32 KB (W1 -> W2, via TDM)
    __shared__ half_t sA[4][16 * 128]; // 16 KB (h tile -> t1 tile)

    const int tid = threadIdx.x, lane = tid & 31, wave = tid >> 5;
    const int hf = lane >> 4, ncol = lane & 15;
    const int rowbase = blockIdx.x * 64 + wave * 16;

    if (wave == 0) tdm_load_1d(W1t, sW, 8192);
    { const float4* h4 = (const float4*)hin;
#pragma unroll
      for (int it = 0; it < 16; ++it) {
          int fidx = it * 32 + lane;
          int r = fidx >> 5, c4 = fidx & 31;
          int g = rowbase + r;
          float4 v = (g < E_) ? h4[(size_t)g * 32 + c4] : make_float4(0.f, 0.f, 0.f, 0.f);
          union { half_t h[4]; uint2 u; } pk;
          pk.h[0] = (half_t)v.x; pk.h[1] = (half_t)v.y;
          pk.h[2] = (half_t)v.z; pk.h[3] = (half_t)v.w;
          *(uint2*)&sA[wave][r * 128 + c4 * 4] = pk.u;
      } }
    if (wave == 0) tdm_drain();
    __syncthreads();

    v16h a[4];
#pragma unroll
    for (int kb = 0; kb < 4; ++kb) a[kb] = load_a_frag(sA[wave], 128, lane, kb);
    v8f acc[8] = {};
#pragma unroll
    for (int kb = 0; kb < 4; ++kb)
#pragma unroll
        for (int j = 0; j < 8; ++j)
            acc[j] = wmma_f16(a[kb], load_b_frag(sW, 128, j, kb, lane), acc[j]);

#pragma unroll
    for (int j = 0; j < 8; ++j) {
        const int n = j * 16 + ncol;
        const float bn = b1[n];
#pragma unroll
        for (int i = 0; i < 8; ++i) {
            const int m = i + 8 * hf;
            sA[wave][m * 128 + n] = (half_t)silu_f(acc[j][i] + bn);
        }
    }
    __syncthreads();
    if (wave == 0) { tdm_load_1d(W2t, sW, 8192); tdm_drain(); }
    __syncthreads();

#pragma unroll
    for (int kb = 0; kb < 4; ++kb) a[kb] = load_a_frag(sA[wave], 128, lane, kb);
    v8f acc2[8] = {};
#pragma unroll
    for (int kb = 0; kb < 4; ++kb)
#pragma unroll
        for (int j = 0; j < 8; ++j)
            acc2[j] = wmma_f16(a[kb], load_b_frag(sW, 128, j, kb, lane), acc2[j]);

#pragma unroll
    for (int j = 0; j < 8; ++j) {
        const int n = j * 16 + ncol;
        const float bn = b2[n];
#pragma unroll
        for (int i = 0; i < 8; ++i) {
            int g = rowbase + i + 8 * hf;
            if (g < E_)
                hout[(size_t)g * 128 + n] =
                    hin[(size_t)g * 128 + n] + silu_f(acc2[j][i] + bn);
        }
    }
}

// ---------------------------------------------------------------------------
// Linear: hout = silu(hin@Wlin+blin) + xres
// ---------------------------------------------------------------------------
__global__ void __launch_bounds__(128) k_lin(
    const float* __restrict__ hin, const half_t* __restrict__ Wlint,
    const float* __restrict__ blin, const float* __restrict__ xres,
    float* __restrict__ hout) {
    __shared__ half_t sW[128 * 128];
    __shared__ half_t sA[4][16 * 128];

    const int tid = threadIdx.x, lane = tid & 31, wave = tid >> 5;
    const int hf = lane >> 4, ncol = lane & 15;
    const int rowbase = blockIdx.x * 64 + wave * 16;

    if (wave == 0) tdm_load_1d(Wlint, sW, 8192);
    { const float4* h4 = (const float4*)hin;
#pragma unroll
      for (int it = 0; it < 16; ++it) {
          int fidx = it * 32 + lane;
          int r = fidx >> 5, c4 = fidx & 31;
          int g = rowbase + r;
          float4 v = (g < E_) ? h4[(size_t)g * 32 + c4] : make_float4(0.f, 0.f, 0.f, 0.f);
          union { half_t h[4]; uint2 u; } pk;
          pk.h[0] = (half_t)v.x; pk.h[1] = (half_t)v.y;
          pk.h[2] = (half_t)v.z; pk.h[3] = (half_t)v.w;
          *(uint2*)&sA[wave][r * 128 + c4 * 4] = pk.u;
      } }
    if (wave == 0) tdm_drain();
    __syncthreads();

    v16h a[4];
#pragma unroll
    for (int kb = 0; kb < 4; ++kb) a[kb] = load_a_frag(sA[wave], 128, lane, kb);
    v8f acc[8] = {};
#pragma unroll
    for (int kb = 0; kb < 4; ++kb)
#pragma unroll
        for (int j = 0; j < 8; ++j)
            acc[j] = wmma_f16(a[kb], load_b_frag(sW, 128, j, kb, lane), acc[j]);

#pragma unroll
    for (int j = 0; j < 8; ++j) {
        const int n = j * 16 + ncol;
        const float bn = blin[n];
#pragma unroll
        for (int i = 0; i < 8; ++i) {
            int g = rowbase + i + 8 * hf;
            if (g < E_)
                hout[(size_t)g * 128 + n] =
                    silu_f(acc[j][i] + bn) + xres[(size_t)g * 128 + n];
        }
    }
}

// ---------------------------------------------------------------------------
extern "C" void kernel_launch(void* const* d_in, const int* in_sizes, int n_in,
                              void* d_out, int out_size, void* d_ws, size_t ws_size,
                              hipStream_t stream) {
    const float* x      = (const float*)d_in[0];
    const float* rbf    = (const float*)d_in[1];
    const float* sbf    = (const float*)d_in[2];
    const int*   idx_kj = (const int*)d_in[3];
    const int*   idx_ji = (const int*)d_in[4];
    const float* Wrbf1  = (const float*)d_in[5];
    const float* Wrbf2  = (const float*)d_in[6];
    const float* Wsbf1  = (const float*)d_in[7];
    const float* Wsbf2  = (const float*)d_in[8];
    const float* Wkj    = (const float*)d_in[9];
    const float* bkj    = (const float*)d_in[10];
    const float* Wji    = (const float*)d_in[11];
    const float* bji    = (const float*)d_in[12];
    const float* Wdown  = (const float*)d_in[13];
    const float* Wup    = (const float*)d_in[14];
    const float* Wlin   = (const float*)d_in[15];
    const float* blin   = (const float*)d_in[16];
    const float* W1b    = (const float*)d_in[17];
    const float* b1b    = (const float*)d_in[18];
    const float* W2b    = (const float*)d_in[19];
    const float* b2b    = (const float*)d_in[20];
    const float* W1a0   = (const float*)d_in[21];
    const float* b1a0   = (const float*)d_in[22];
    const float* W2a0   = (const float*)d_in[23];
    const float* b2a0   = (const float*)d_in[24];
    const float* W1a1   = (const float*)d_in[25];
    const float* b1a1   = (const float*)d_in[26];
    const float* W2a1   = (const float*)d_in[27];
    const float* b2a1   = (const float*)d_in[28];

    char* ws = (char*)d_ws;
    float*  xji   = (float*)(ws + OFF_XJI);
    float*  xkd   = (float*)(ws + OFF_XKD);
    float*  ssum  = (float*)(ws + OFF_SSUM);
    float*  cnt   = (float*)(ws + OFF_CNT);
    float*  hbuf  = (float*)(ws + OFF_HBUF);
    float*  wr    = (float*)(ws + OFF_WR);
    half_t* Wjit  = (half_t*)(ws + OFF_WJIT);
    half_t* Wkjt  = (half_t*)(ws + OFF_WKJT);
    half_t* Wlint = (half_t*)(ws + OFF_WLINT);
    half_t* W1bt  = (half_t*)(ws + OFF_W1BT);
    half_t* W2bt  = (half_t*)(ws + OFF_W2BT);
    half_t* W1a0t = (half_t*)(ws + OFF_W1A0T);
    half_t* W2a0t = (half_t*)(ws + OFF_W2A0T);
    half_t* W1a1t = (half_t*)(ws + OFF_W1A1T);
    half_t* W2a1t = (half_t*)(ws + OFF_W2A1T);
    half_t* Wdownt= (half_t*)(ws + OFF_WDOWNT);
    half_t* Wupt  = (half_t*)(ws + OFF_WUPT);
    half_t* Wsbft = (half_t*)(ws + OFF_WSBFT);
    float* out = (float*)d_out;

    const int EGRID = (E_ + 63) / 64;   // 4688
    const int TGRID = T_ / 64;          // 46875

    // ---- weight prep ----
    k_wr  <<<1, 256, 0, stream>>>(Wrbf1, Wrbf2, wr);
    k_wsbf<<<16, 256, 0, stream>>>(Wsbf1, Wsbf2, Wsbft);
    k_transpose<<<64, 256, 0, stream>>>(Wji,  Wjit,  128, 128);
    k_transpose<<<64, 256, 0, stream>>>(Wkj,  Wkjt,  128, 128);
    k_transpose<<<64, 256, 0, stream>>>(Wlin, Wlint, 128, 128);
    k_transpose<<<64, 256, 0, stream>>>(W1b,  W1bt,  128, 128);
    k_transpose<<<64, 256, 0, stream>>>(W2b,  W2bt,  128, 128);
    k_transpose<<<64, 256, 0, stream>>>(W1a0, W1a0t, 128, 128);
    k_transpose<<<64, 256, 0, stream>>>(W2a0, W2a0t, 128, 128);
    k_transpose<<<64, 256, 0, stream>>>(W1a1, W1a1t, 128, 128);
    k_transpose<<<64, 256, 0, stream>>>(W2a1, W2a1t, 128, 128);
    k_transpose<<<32, 256, 0, stream>>>(Wdown, Wdownt, 128, 64);
    k_transpose<<<32, 256, 0, stream>>>(Wup,   Wupt,   64, 128);

    // ssum + cnt are contiguous -> zero in one pass (E*64 + E floats)
    k_zero<<<8192, 256, 0, stream>>>(ssum, (long long)E_ * 64 + E_);

    // ---- main pipeline ----
    k_edge   <<<EGRID, 128, 0, stream>>>(x, rbf, Wjit, Wkjt, Wdownt, wr,
                                         bji, bkj, xji, xkd);
    k_triplet<<<TGRID, 128, 0, stream>>>(sbf, idx_kj, idx_ji, Wsbft, xkd,
                                         ssum, cnt);
    k_up     <<<EGRID, 128, 0, stream>>>(ssum, cnt, Wupt, xji, out);
    k_res    <<<EGRID, 128, 0, stream>>>(out,  W1bt, b1b, W2bt, b2b, hbuf);
    k_lin    <<<EGRID, 128, 0, stream>>>(hbuf, Wlint, blin, x, out);
    k_res    <<<EGRID, 128, 0, stream>>>(out,  W1a0t, b1a0, W2a0t, b2a0, hbuf);
    k_res    <<<EGRID, 128, 0, stream>>>(hbuf, W1a1t, b1a1, W2a1t, b2a1, out);
}
